// LevelDepthAwareCrossAttention_48215302865368
// MI455X (gfx1250) — matrix operand order repaired and checked by reading, not compiled, not run on hardware
//
#include <hip/hip_runtime.h>
#include <hip/hip_bf16.h>

typedef __attribute__((ext_vector_type(16))) _Float16 v16h;
typedef __attribute__((ext_vector_type(8)))  _Float16 v8h;
typedef __attribute__((ext_vector_type(8)))  float    v8f;
typedef __attribute__((ext_vector_type(4)))  int      v4i;

#define DEVINL __device__ __forceinline__
#define AS1 __attribute__((address_space(1)))
#define AS3 __attribute__((address_space(3)))

DEVINL v8f wmma_f16(v16h a, v16h b, v8f c) {
  // D = A(16x32 f16) * B(32x16 f16) + C(16x16 f32), wave32
  return __builtin_amdgcn_wmma_f32_16x16x32_f16(false, a, false, b, (short)0, c, false, false);
}

DEVINL void wait_async_zero() {
#if __has_builtin(__builtin_amdgcn_s_wait_asynccnt)
  __builtin_amdgcn_s_wait_asynccnt(0);
#else
  asm volatile("s_wait_asynccnt 0" ::: "memory");
#endif
}

DEVINL v16h cat16(v8h lo, v8h hi) {
  v16h a;
#pragma unroll
  for (int e = 0; e < 8; ++e) { a[e] = lo[e]; a[e + 8] = hi[e]; }
  return a;
}

// ---- WMMA fragment loaders (wave32 16x16x32 f16; contiguous k order, same in A and B) ----
// A (16x32): lane holds row M=lane&15. lanes<16: K k0..k0+15;
//            lanes>=16: K k0+16..k0+31. Contiguous 16-elt halves.
DEVINL v16h loadA_vec(const _Float16* src, int ld, int m0, int k0, int lane, int maxRow) {
  const int m = m0 + (lane & 15);
  const int kh = k0 + ((lane & 16) ? 16 : 0);
  if (m < maxRow) {
    v8h lo = *(const v8h*)(src + m * ld + kh);
    v8h hi = *(const v8h*)(src + m * ld + kh + 8);
    return cat16(lo, hi);
  }
  v16h z = {};
  return z;
}

DEVINL v16h loadA_guard(const _Float16* src, int ld, int m0, int k0, int lane,
                        int maxRow, int maxK) {
  const int m = m0 + (lane & 15);
  const int kh = k0 + ((lane & 16) ? 16 : 0);
  v16h a = {};
  if (m < maxRow) {
#pragma unroll
    for (int e = 0; e < 16; ++e) {
      const int k = kh + e;
      a[e] = (k < maxK) ? src[m * ld + k] : (_Float16)0.0f;
    }
  }
  return a;
}

// B (32x16) built as W^T from row-major W[N][K]: B[k][n] = W[n][k].
// lane holds col N=lane&15. lanes<16: K k0..k0+15; lanes>=16: K k0+16..k0+31.
DEVINL v16h loadBT_vec(const _Float16* W, int ldw, int n0, int k0, int lane, int maxN) {
  const int n = n0 + (lane & 15);
  const int kb = k0 + ((lane & 16) ? 16 : 0);
  if (n < maxN) {
    v8h lo = *(const v8h*)(W + n * ldw + kb);
    v8h hi = *(const v8h*)(W + n * ldw + kb + 8);
    return cat16(lo, hi);
  }
  v16h z = {};
  return z;
}

DEVINL v16h loadBT_guard(const _Float16* W, int ldw, int n0, int k0, int lane,
                         int maxN, int maxK) {
  const int n = n0 + (lane & 15);
  const int kb = k0 + ((lane & 16) ? 16 : 0);
  v16h b = {};
  if (n < maxN) {
#pragma unroll
    for (int e = 0; e < 16; ++e) {
      const int k = kb + e;
      b[e] = (k < maxK) ? W[n * ldw + k] : (_Float16)0.0f;
    }
  }
  return b;
}

// B direct from row-major V[K][N] (used for att @ V).
DEVINL v16h loadB_dir(const _Float16* V, int ld, int k0, int n0, int lane,
                      int maxK, int maxN) {
  const int n = n0 + (lane & 15);
  const int kb = k0 + ((lane & 16) ? 16 : 0);
  v16h b = {};
  if (n < maxN) {
#pragma unroll
    for (int e = 0; e < 16; ++e) {
      const int k = kb + e;
      b[e] = (k < maxK) ? V[k * ld + n] : (_Float16)0.0f;
    }
  }
  return b;
}

// C/D (16x16 f32): VGPR r holds row r (+8 for lanes>=16), col = lane&15.
DEVINL void storeD_f16(_Float16* dst, int ld, int m0, int n0, int lane, v8f c,
                       int maxRow, const float* bias) {
  const int n = n0 + (lane & 15);
  const int mb = m0 + ((lane & 16) ? 8 : 0);
  const float bvn = bias[n];
#pragma unroll
  for (int r = 0; r < 8; ++r) {
    const int m = mb + r;
    if (m < maxRow) dst[m * ld + n] = (_Float16)(c[r] + bvn);
  }
}

// ---------------------------------------------------------------------------
// Prep: effective fused weights Wq=wi_q@qw, Wk=wi_k@kw, Wv=wi_v@vw (f16),
// wo->f16, effective biases b*_eff = wi_*@b* + bi_*.
// ---------------------------------------------------------------------------
template <int L>
__global__ void prep_kernel(const float* __restrict__ qw, const float* __restrict__ kw,
                            const float* __restrict__ vw, const float* __restrict__ qb,
                            const float* __restrict__ kb, const float* __restrict__ vb,
                            const float* __restrict__ wi, const float* __restrict__ bi,
                            const float* __restrict__ wo, _Float16* __restrict__ Wq,
                            _Float16* __restrict__ Wk, _Float16* __restrict__ Wv,
                            _Float16* __restrict__ Wo16, float* __restrict__ bq,
                            float* __restrict__ bk, float* __restrict__ bv) {
  constexpr int C1 = 64 << L;
  const int idx = blockIdx.x * blockDim.x + threadIdx.x;
  const int nW = 3 * C1 * C1;
  if (idx < nW) {
    const int which = idx / (C1 * C1);
    const int rc = idx % (C1 * C1);
    const int r = rc / C1, c = rc % C1;
    const float* M = (which == 0) ? qw : ((which == 1) ? kw : vw);
    const float* wiw = wi + (size_t)which * C1 * C1;
    float s = 0.0f;
    for (int k = 0; k < C1; ++k) s += wiw[r * C1 + k] * M[k * C1 + c];
    _Float16* dst = (which == 0) ? Wq : ((which == 1) ? Wk : Wv);
    dst[rc] = (_Float16)s;
  } else if (idx < nW + C1 * C1) {
    const int rc = idx - nW;
    Wo16[rc] = (_Float16)wo[rc];
  } else if (idx < nW + C1 * C1 + 3 * C1) {
    const int t = idx - nW - C1 * C1;
    const int which = t / C1, r = t % C1;
    const float* bvec = (which == 0) ? qb : ((which == 1) ? kb : vb);
    const float* wiw = wi + (size_t)which * C1 * C1;
    float s = bi[which * C1 + r];
    for (int k = 0; k < C1; ++k) s += wiw[r * C1 + k] * bvec[k];
    float* dst = (which == 0) ? bq : ((which == 1) ? bk : bv);
    dst[r] = s;
  }
}

// ---------------------------------------------------------------------------
// Canvas: polar bilinear gather from BEV + positional embed -> qseq f16
// layout [g][j][i][c].
// ---------------------------------------------------------------------------
template <int L>
__global__ __launch_bounds__(256) void canvas_kernel(
    const float* __restrict__ a, const float* __restrict__ pa,
    const float* __restrict__ fovs, const float* __restrict__ rots,
    _Float16* __restrict__ qseq) {
  constexpr int C1 = 64 << L, H1 = 128 >> L, W2 = 240 >> L, HW = 256 >> L;
  const int j = blockIdx.x, g = blockIdx.y;
  const float fov = fovs[g];
  const float* R = rots + g * 9;
  const float cr = -R[3], sr = R[0];  // r = [[0,-1],[1,0]] @ rot[:2,:2]
  const float ang = -0.5f * fov + fov * (float)j / (float)(W2 - 1);
  const float ca = cr * cosf(ang) + sr * sinf(ang);
  const float sa = -sr * cosf(ang) + cr * sinf(ang);
  const float cxp = (float)(HW / 2), cyp = (float)(HW / 2);
  const float rmax = sqrtf(cxp * cxp + cyp * cyp);
  const float* ag = a + (size_t)g * C1 * HW * HW;
  _Float16* dst = qseq + (size_t)(g * W2 + j) * H1 * C1;
  for (int t = threadIdx.x; t < H1 * C1; t += blockDim.x) {
    const int i = t % H1, c = t / H1;
    const float rad = rmax * (float)i / (float)(H1 - 1);
    float x = cxp + rad * ca, y = cyp - rad * sa;
    x = fminf(fmaxf(x, 0.0f), (float)(HW - 1));
    y = fminf(fmaxf(y, 0.0f), (float)(HW - 1));
    const float x0f = floorf(x), y0f = floorf(y);
    const float wx = x - x0f, wy = y - y0f;
    const int x0 = min(max((int)x0f, 0), HW - 1), x1 = min(x0 + 1, HW - 1);
    const int y0 = min(max((int)y0f, 0), HW - 1), y1 = min(y0 + 1, HW - 1);
    const float* ac = ag + (size_t)c * HW * HW;
    const float v00 = ac[y0 * HW + x0], v01 = ac[y0 * HW + x1];
    const float v10 = ac[y1 * HW + x0], v11 = ac[y1 * HW + x1];
    const float v = (1.0f - wy) * ((1.0f - wx) * v00 + wx * v01) +
                    wy * ((1.0f - wx) * v10 + wx * v11);
    dst[i * C1 + c] = (_Float16)(v + pa[i * C1 + c]);
  }
}

// ---------------------------------------------------------------------------
// bseq: b (g,c,kk,j) transpose + embed -> f16 [g][j][kk][c].
// ---------------------------------------------------------------------------
template <int L>
__global__ void bseq_kernel(const float* __restrict__ b, const float* __restrict__ pb,
                            _Float16* __restrict__ bseq) {
  constexpr int C1 = 64 << L, H2 = 48 >> L, W2 = 240 >> L;
  const size_t n = (size_t)4 * C1 * H2 * W2;
  for (size_t idx = (size_t)blockIdx.x * blockDim.x + threadIdx.x; idx < n;
       idx += (size_t)gridDim.x * blockDim.x) {
    const int j = (int)(idx % W2);
    const int kk = (int)((idx / W2) % H2);
    const int c = (int)((idx / ((size_t)W2 * H2)) % C1);
    const int g = (int)(idx / ((size_t)W2 * H2 * C1));
    const float v = b[idx] + pb[kk * C1 + c];
    bseq[(((size_t)g * W2 + j) * H2 + kk) * C1 + c] = (_Float16)v;
  }
}

// ---------------------------------------------------------------------------
// Fused per-column attention: one workgroup per (column j, agent g).
// 8 wave32 waves; all GEMMs via v_wmma_f32_16x16x32_f16.
// Column staging via GLOBAL_LOAD_ASYNC_TO_LDS (ASYNCcnt) when available.
// Epilogue: scatter-add (restore_canvas numerator) + cnt.
// ---------------------------------------------------------------------------
template <int L>
__global__ __launch_bounds__(256) void attn_kernel(
    const _Float16* __restrict__ qseq, const _Float16* __restrict__ bseq,
    const _Float16* __restrict__ Wq, const _Float16* __restrict__ Wk,
    const _Float16* __restrict__ Wv, const _Float16* __restrict__ Wo,
    const float* __restrict__ bq, const float* __restrict__ bk,
    const float* __restrict__ bv, const float* __restrict__ bo,
    const float* __restrict__ fovs, const float* __restrict__ rots,
    float* __restrict__ outAcc, float* __restrict__ cnt) {
  constexpr int C1 = 64 << L, H1 = 128 >> L, H2 = 48 >> L, W2 = 240 >> L;
  constexpr int HW = 256 >> L, HD = 8 << L;
  constexpr int H2P = (H2 + 31) & ~31;  // K-pad for att@V
  constexpr int NW = 8;                 // waves per block (wave32)
  const int j = blockIdx.x, g = blockIdx.y;
  const int tid = threadIdx.x, lane = tid & 31, wid = tid >> 5;

  __shared__ alignas(16) _Float16 sQ[H1 * C1];  // reused as sO after projections
  __shared__ alignas(16) _Float16 sB[H2 * C1];
  __shared__ alignas(16) _Float16 sQP[H1 * C1];
  __shared__ alignas(16) _Float16 sKP[H2 * C1];
  __shared__ alignas(16) _Float16 sVP[H2 * C1];
  __shared__ alignas(16) _Float16 sA[H1 * H2P];
  __shared__ alignas(16) float sS[H1 * H2];
  __shared__ int sIdx[H1];
  _Float16* sO = sQ;

  // Warm L2/WGP$ for the shared weight matrices (global_prefetch_b8).
  if (tid == 0) {
    __builtin_prefetch(Wq, 0, 0);
    __builtin_prefetch(Wk, 0, 0);
    __builtin_prefetch(Wv, 0, 0);
    __builtin_prefetch(Wo, 0, 0);
  }

  {  // stage this column's sequences into LDS
    const _Float16* srcq = qseq + (size_t)(g * W2 + j) * H1 * C1;
    const _Float16* srcb = bseq + (size_t)(g * W2 + j) * H2 * C1;
#if __has_builtin(__builtin_amdgcn_global_load_async_to_lds_b128)
    // DMA path: 16B per lane per issue, no VGPR round-trip (ASYNCcnt).
    for (int t = tid * 8; t < H1 * C1; t += 256 * 8)
      __builtin_amdgcn_global_load_async_to_lds_b128(
          (AS1 v4i*)(srcq + t), (AS3 v4i*)(sQ + t), 0, 0);
    for (int t = tid * 8; t < H2 * C1; t += 256 * 8)
      __builtin_amdgcn_global_load_async_to_lds_b128(
          (AS1 v4i*)(srcb + t), (AS3 v4i*)(sB + t), 0, 0);
#else
    for (int t = tid; t < H1 * C1; t += 256) sQ[t] = srcq[t];
    for (int t = tid; t < H2 * C1; t += 256) sB[t] = srcb[t];
#endif
  }
  if (tid < H1) {  // scatter indices (restore_canvas rounding) + cnt
    const float fov = fovs[g];
    const float* R = rots + g * 9;
    const float cr = -R[3], sr = R[0];
    const float ang = -0.5f * fov + fov * (float)j / (float)(W2 - 1);
    const float ca = cr * cosf(ang) + sr * sinf(ang);
    const float sa = -sr * cosf(ang) + cr * sinf(ang);
    const float cxp = (float)(HW / 2), cyp = (float)(HW / 2);
    const float rmax = sqrtf(cxp * cxp + cyp * cyp);
    const float rad = rmax * (float)tid / (float)(H1 - 1);
    const float x = cxp + rad * ca;
    const float y = cyp - rad * sa;
    const int xi = min(max((int)rintf(x), 0), HW - 1);
    const int yi = min(max((int)rintf(y), 0), HW - 1);
    sIdx[tid] = yi * HW + xi;
    atomicAdd(&cnt[(size_t)g * HW * HW + sIdx[tid]], 1.0f);
  }
#if __has_builtin(__builtin_amdgcn_global_load_async_to_lds_b128)
  wait_async_zero();
#endif
  __syncthreads();

  {  // QP = sQ @ Wq^T + bq  (M=H1, N=C1, K=C1)
    constexpr int MT = H1 / 16, NT = C1 / 16, KS = C1 / 32;
    for (int t = wid; t < MT * NT; t += NW) {
      const int mt = t / NT, nt = t % NT;
      v8f acc = {};
      for (int ks = 0; ks < KS; ++ks) {
        v16h af = loadA_vec(sQ, C1, mt * 16, ks * 32, lane, H1);
        v16h bf = loadBT_vec(Wq, C1, nt * 16, ks * 32, lane, C1);
        acc = wmma_f16(af, bf, acc);
      }
      storeD_f16(sQP, C1, mt * 16, nt * 16, lane, acc, H1, bq);
    }
  }
  {  // KP/VP = sB @ W{k,v}^T + b  (M=H2, N=C1, K=C1)
    constexpr int MT = (H2 + 15) / 16, NT = C1 / 16, KS = C1 / 32;
    for (int t = wid; t < 2 * MT * NT; t += NW) {
      const int which = t / (MT * NT);
      const int tt = t % (MT * NT);
      const int mt = tt / NT, nt = tt % NT;
      const _Float16* W = which ? Wv : Wk;
      const float* bias = which ? bv : bk;
      _Float16* dst = which ? sVP : sKP;
      v8f acc = {};
      for (int ks = 0; ks < KS; ++ks) {
        v16h af = loadA_vec(sB, C1, mt * 16, ks * 32, lane, H2);
        v16h bf = loadBT_vec(W, C1, nt * 16, ks * 32, lane, C1);
        acc = wmma_f16(af, bf, acc);
      }
      storeD_f16(dst, C1, mt * 16, nt * 16, lane, acc, H2, bias);
    }
  }
  __syncthreads();

  const float scale = 1.0f / sqrtf((float)HD);
  for (int h = 0; h < 8; ++h) {
    {  // scores = qp_h @ kp_h^T  (K = HD, zero-padded to 32)
      constexpr int MT = H1 / 16, NT = (H2 + 15) / 16;
      for (int t = wid; t < MT * NT; t += NW) {
        const int mt = t / NT, nt = t % NT;
        v16h af = loadA_guard(sQP + h * HD, C1, mt * 16, 0, lane, H1, HD);
        v16h bf = loadBT_guard(sKP + h * HD, C1, nt * 16, 0, lane, H2, HD);
        v8f acc = {};
        acc = wmma_f16(af, bf, acc);
        const int n = nt * 16 + (lane & 15);
        const int mb = mt * 16 + ((lane & 16) ? 8 : 0);
        if (n < H2) {
#pragma unroll
          for (int r = 0; r < 8; ++r) sS[(mb + r) * H2 + n] = acc[r];
        }
      }
    }
    __syncthreads();
    if (tid < H1) {  // softmax over kk (scale applied pre-softmax)
      float m = -3.0e38f;
      for (int k2 = 0; k2 < H2; ++k2) m = fmaxf(m, sS[tid * H2 + k2] * scale);
      float s = 0.0f;
      for (int k2 = 0; k2 < H2; ++k2) s += __expf(sS[tid * H2 + k2] * scale - m);
      const float inv = 1.0f / s;
      for (int k2 = 0; k2 < H2P; ++k2)
        sA[tid * H2P + k2] = (k2 < H2)
            ? (_Float16)(__expf(sS[tid * H2 + k2] * scale - m) * inv)
            : (_Float16)0.0f;
    }
    __syncthreads();
    {  // o_h = att @ vp_h  (M=H1, N=HD, K=H2 padded to H2P)
      constexpr int MT = H1 / 16, NT = (HD + 15) / 16, KS = H2P / 32;
      for (int t = wid; t < MT * NT; t += NW) {
        const int mt = t / NT, nt = t % NT;
        v8f acc = {};
        for (int ks = 0; ks < KS; ++ks) {
          v16h af = loadA_vec(sA, H2P, mt * 16, ks * 32, lane, H1);
          v16h bf = loadB_dir(sVP + h * HD, C1, ks * 32, nt * 16, lane, H2, HD);
          acc = wmma_f16(af, bf, acc);
        }
        const int n = nt * 16 + (lane & 15);
        const int mb = mt * 16 + ((lane & 16) ? 8 : 0);
        if (n < HD) {
#pragma unroll
          for (int r = 0; r < 8; ++r)
            sO[(mb + r) * C1 + h * HD + n] = (_Float16)acc[r];
        }
      }
    }
    __syncthreads();
  }

  {  // out = o @ wo^T + bo; scatter-add into BEV accumulator (d_out)
    constexpr int MT = H1 / 16, NT = C1 / 16, KS = C1 / 32;
    for (int t = wid; t < MT * NT; t += NW) {
      const int mt = t / NT, nt = t % NT;
      v8f acc = {};
      for (int ks = 0; ks < KS; ++ks) {
        v16h af = loadA_vec(sO, C1, mt * 16, ks * 32, lane, H1);
        v16h bf = loadBT_vec(Wo, C1, nt * 16, ks * 32, lane, C1);
        acc = wmma_f16(af, bf, acc);
      }
      const int n = nt * 16 + (lane & 15);
      const float bvn = bo[n];
      const int mb = mt * 16 + ((lane & 16) ? 8 : 0);
#pragma unroll
      for (int r = 0; r < 8; ++r) {
        const int i = mb + r;
        atomicAdd(&outAcc[((size_t)g * C1 + n) * (HW * HW) + sIdx[i]], acc[r] + bvn);
      }
    }
  }
}

// ---------------------------------------------------------------------------
// Finalize: out = a + acc / max(cnt,1)  (memory-bound, coalesced)
// ---------------------------------------------------------------------------
template <int L>
__global__ void finalize_kernel(const float* __restrict__ a, const float* __restrict__ cnt,
                                float* __restrict__ out) {
  constexpr int C1 = 64 << L, HW = 256 >> L;
  const size_t HW2 = (size_t)HW * HW;
  const size_t n = (size_t)4 * C1 * HW2;
  for (size_t idx = (size_t)blockIdx.x * blockDim.x + threadIdx.x; idx < n;
       idx += (size_t)gridDim.x * blockDim.x) {
    const size_t pix = idx % HW2;
    const size_t g = idx / ((size_t)C1 * HW2);
    float c = cnt[g * HW2 + pix];
    if (c == 0.0f) c = 1.0f;
    out[idx] = a[idx] + out[idx] / c;
  }
}

// ---------------------------------------------------------------------------
template <int L>
static void run_level(void* const* d_in, float* outL, _Float16* qseq, _Float16* bseq,
                      float* cnt, _Float16* Wq, _Float16* Wk, _Float16* Wv, _Float16* Wo,
                      float* bq, float* bk, float* bv, hipStream_t stream) {
  constexpr int C1 = 64 << L, H2 = 48 >> L, W2 = 240 >> L, HW = 256 >> L;
  const float* a = (const float*)d_in[L];
  const float* b = (const float*)d_in[3 + L];
  const float* fovs = (const float*)d_in[6];
  const float* rots = (const float*)d_in[7];
  const int pbase = 11 + 12 * L;
  const float* qw = (const float*)d_in[pbase + 0];
  const float* qb = (const float*)d_in[pbase + 1];
  const float* kw = (const float*)d_in[pbase + 2];
  const float* kb = (const float*)d_in[pbase + 3];
  const float* vw = (const float*)d_in[pbase + 4];
  const float* vb = (const float*)d_in[pbase + 5];
  const float* pa = (const float*)d_in[pbase + 6];
  const float* pbp = (const float*)d_in[pbase + 7];
  const float* wi = (const float*)d_in[pbase + 8];
  const float* bi = (const float*)d_in[pbase + 9];
  const float* wo = (const float*)d_in[pbase + 10];
  const float* bo = (const float*)d_in[pbase + 11];

  (void)hipMemsetAsync(cnt, 0, (size_t)4 * HW * HW * sizeof(float), stream);
  const int nprep = 4 * C1 * C1 + 3 * C1;
  prep_kernel<L><<<(nprep + 255) / 256, 256, 0, stream>>>(qw, kw, vw, qb, kb, vb, wi, bi,
                                                          wo, Wq, Wk, Wv, Wo, bq, bk, bv);
  canvas_kernel<L><<<dim3(W2, 4), 256, 0, stream>>>(a, pa, fovs, rots, qseq);
  const size_t nb = (size_t)4 * C1 * H2 * W2;
  bseq_kernel<L><<<(int)((nb + 255) / 256), 256, 0, stream>>>(b, pbp, bseq);
  attn_kernel<L><<<dim3(W2, 4), 256, 0, stream>>>(qseq, bseq, Wq, Wk, Wv, Wo, bq, bk, bv,
                                                  bo, fovs, rots, outL, cnt);
  finalize_kernel<L><<<4096, 256, 0, stream>>>(a, cnt, outL);
}

extern "C" void kernel_launch(void* const* d_in, const int* in_sizes, int n_in,
                              void* d_out, int out_size, void* d_ws, size_t ws_size,
                              hipStream_t stream) {
  (void)in_sizes; (void)n_in; (void)ws_size;
  float* out = (float*)d_out;
  (void)hipMemsetAsync(d_out, 0, (size_t)out_size * sizeof(float), stream);

  char* ws = (char*)d_ws;
  size_t off = 0;
  auto carve = [&](size_t bytes) -> void* {
    void* p = (void*)(ws + off);
    off = (off + bytes + 255) & ~(size_t)255;
    return p;
  };
  _Float16* qseq = (_Float16*)carve((size_t)4 * 240 * 128 * 64 * sizeof(_Float16));
  _Float16* bseq = (_Float16*)carve((size_t)4 * 240 * 48 * 64 * sizeof(_Float16));
  float* cnt = (float*)carve((size_t)4 * 256 * 256 * sizeof(float));
  _Float16* Wq = (_Float16*)carve((size_t)256 * 256 * sizeof(_Float16));
  _Float16* Wk = (_Float16*)carve((size_t)256 * 256 * sizeof(_Float16));
  _Float16* Wv = (_Float16*)carve((size_t)256 * 256 * sizeof(_Float16));
  _Float16* Wo = (_Float16*)carve((size_t)256 * 256 * sizeof(_Float16));
  float* bq = (float*)carve(256 * sizeof(float));
  float* bk = (float*)carve(256 * sizeof(float));
  float* bv = (float*)carve(256 * sizeof(float));

  const size_t o0 = 0;
  const size_t o1 = o0 + (size_t)4 * 64 * 256 * 256;
  const size_t o2 = o1 + (size_t)4 * 128 * 128 * 128;
  run_level<0>(d_in, out + o0, qseq, bseq, cnt, Wq, Wk, Wv, Wo, bq, bk, bv, stream);
  run_level<1>(d_in, out + o1, qseq, bseq, cnt, Wq, Wk, Wv, Wo, bq, bk, bv, stream);
  run_level<2>(d_in, out + o2, qseq, bseq, cnt, Wq, Wk, Wv, Wo, bq, bk, bv, stream);
}
